// DeepseekAttention_52570399703712
// MI455X (gfx1250) — matrix operand order repaired and checked by reading, not compiled, not run on hardware
//
#include <hip/hip_runtime.h>
#include <cstddef>
#include <cstdint>

typedef __attribute__((ext_vector_type(4)))  float    v4f;
typedef __attribute__((ext_vector_type(4)))  _Float16 v4h;
typedef __attribute__((ext_vector_type(8)))  _Float16 v8h;
typedef __attribute__((ext_vector_type(16))) _Float16 v16h;
typedef __attribute__((ext_vector_type(8)))  float    v8f;

#define TT   4096
#define HID  2048
#define NH   16
#define HD   128

__device__ __forceinline__ v16h pk16(v8h a, v8h b) {
    return __builtin_shufflevector(a, b, 0,1,2,3,4,5,6,7,8,9,10,11,12,13,14,15);
}

__device__ __forceinline__ v8f wmma_f16(v16h a, v16h b, v8f c) {
    // D = A(16x32 f16) * B(32x16 f16) + C(16x16 f32)
    return __builtin_amdgcn_wmma_f32_16x16x32_f16(false, a, false, b, (short)0, c, false, false);
}

// CDNA5 async copy: per-lane 16B global -> LDS, tracked by ASYNCcnt.
__device__ __forceinline__ void async_b128(void* lds, const void* g) {
    uint32_t l = (uint32_t)(uintptr_t)lds;   // LDS byte address
    asm volatile("global_load_async_to_lds_b128 %0, %1, off"
                 :: "v"(l), "v"(g) : "memory");
}
__device__ __forceinline__ void wait_async0() {
    asm volatile("s_wait_asynccnt 0x0" ::: "memory");
}

// ---------------------------------------------------------------------------
// fp32 -> f16 streaming convert (vector x4)
// ---------------------------------------------------------------------------
__global__ void __launch_bounds__(256)
cvt_f16_kernel(const float* __restrict__ in, _Float16* __restrict__ out, int n4)
{
    int i = blockIdx.x * blockDim.x + threadIdx.x;
    if (i >= n4) return;
    v4f x = *reinterpret_cast<const v4f*>(in + (size_t)i * 4);
    v4h y;
#pragma unroll
    for (int e = 0; e < 4; e++) y[e] = (_Float16)x[e];
    *reinterpret_cast<v4h*>(out + (size_t)i * 4) = y;
}

// ---------------------------------------------------------------------------
// fp32 [R][C] -> f16 transposed [C][R], LDS-tiled 32x32
// ---------------------------------------------------------------------------
__global__ void __launch_bounds__(256)
tcvt_kernel(const float* __restrict__ in, _Float16* __restrict__ out, int R, int C)
{
    __shared__ float tile[32][33];
    const int tx = threadIdx.x & 31, ty = threadIdx.x >> 5;
    const int c0 = blockIdx.x * 32, r0 = blockIdx.y * 32;
#pragma unroll
    for (int j = 0; j < 4; j++)
        tile[ty + 8 * j][tx] = in[(size_t)(r0 + ty + 8 * j) * C + c0 + tx];
    __syncthreads();
#pragma unroll
    for (int j = 0; j < 4; j++)
        out[(size_t)(c0 + ty + 8 * j) * R + r0 + tx] = (_Float16)tile[tx][ty + 8 * j];
}

// ---------------------------------------------------------------------------
// GEMM: C[MxN] = A16[MxK] * Bt16[NxK]^T (both f16, B pre-transposed, k-contig).
// Tiles staged via async-to-LDS (ASYNCcnt). WMMA f16 x f32-acc.
// mode 0: fp32 C row-major.  mode 1: qkv pack f16 head-major into Qo/Ko/Vo.
// Block 256 = 8 waves (2x4); block tile 64x128; wave tile 32x32.
// ---------------------------------------------------------------------------
#define TILE_M 64
#define TILE_N 128
#define TILE_K 32

__global__ void __launch_bounds__(256)
gemm_wmma_kernel(const _Float16* __restrict__ A16, const _Float16* __restrict__ Bt16,
                 float* __restrict__ Cf,
                 _Float16* __restrict__ Qo, _Float16* __restrict__ Ko,
                 _Float16* __restrict__ Vo,
                 int M, int N, int K, int mode)
{
    __shared__ _Float16 As[TILE_M][TILE_K];   // [m][k]
    __shared__ _Float16 Bs[TILE_N][TILE_K];   // [n][k]

    const int tid  = threadIdx.x;
    const int lane = tid & 31;
    const int wid  = tid >> 5;
    const int l16  = lane & 15;
    const int hi   = lane >> 4;
    const int wm   = wid >> 2;
    const int wn   = wid & 3;
    const int mb   = blockIdx.y * TILE_M;
    const int nb   = blockIdx.x * TILE_N;

    // per-thread 16B staging chunks: A one chunk, B two chunks
    const int ar = tid >> 2, ac = (tid & 3) * 8;
    const _Float16* agp  = A16  + (size_t)(mb + ar) * K + ac;
    const _Float16* bgp0 = Bt16 + (size_t)(nb + ar) * K + ac;
    const _Float16* bgp1 = Bt16 + (size_t)(nb + ar + 64) * K + ac;
    void* alds  = &As[ar][ac];
    void* blds0 = &Bs[ar][ac];
    void* blds1 = &Bs[ar + 64][ac];

    v8f acc[2][2];
#pragma unroll
    for (int i = 0; i < 2; i++)
#pragma unroll
        for (int j = 0; j < 2; j++)
#pragma unroll
            for (int r = 0; r < 8; r++) acc[i][j][r] = 0.0f;

    for (int kb = 0; kb < K; kb += TILE_K) {
        // async-stage this K tile (per-lane 16B copies)
        async_b128(alds,  agp + kb);
        async_b128(blds0, bgp0 + kb);
        async_b128(blds1, bgp1 + kb);
        if (kb + TILE_K < K) {
            __builtin_prefetch(agp + kb + TILE_K, 0, 0);
            __builtin_prefetch(bgp0 + kb + TILE_K, 0, 0);
            __builtin_prefetch(bgp1 + kb + TILE_K, 0, 0);
        }
        wait_async0();
        __syncthreads();

        v16h af[2], bf[2];
#pragma unroll
        for (int sm = 0; sm < 2; sm++) {
            int row = wm * 32 + sm * 16 + l16;
            af[sm] = pk16(*reinterpret_cast<const v8h*>(&As[row][8 * hi]),
                          *reinterpret_cast<const v8h*>(&As[row][16 + 8 * hi]));
        }
#pragma unroll
        for (int sn = 0; sn < 2; sn++) {
            int nrow = wn * 32 + sn * 16 + l16;
            bf[sn] = pk16(*reinterpret_cast<const v8h*>(&Bs[nrow][8 * hi]),
                          *reinterpret_cast<const v8h*>(&Bs[nrow][16 + 8 * hi]));
        }
#pragma unroll
        for (int sm = 0; sm < 2; sm++)
#pragma unroll
            for (int sn = 0; sn < 2; sn++)
                acc[sm][sn] = wmma_f16(af[sm], bf[sn], acc[sm][sn]);
        __syncthreads();
    }

    // writeback: C frag lane = col N=l16, rows r + 8*hi
#pragma unroll
    for (int sm = 0; sm < 2; sm++) {
#pragma unroll
        for (int sn = 0; sn < 2; sn++) {
            int n = nb + wn * 32 + sn * 16 + l16;
#pragma unroll
            for (int r = 0; r < 8; r++) {
                int m = mb + wm * 32 + sm * 16 + 8 * hi + r;
                float v = acc[sm][sn][r];
                if (mode == 0) {
                    Cf[(size_t)m * N + n] = v;
                } else {
                    int which = n >> 11;      // 0=q 1=k 2=v
                    int c2 = n & 2047;
                    int h = c2 >> 7, d = c2 & 127;
                    _Float16* dst = (which == 0) ? Qo : ((which == 1) ? Ko : Vo);
                    dst[(size_t)(h * TT + m) * HD + d] = (_Float16)v;
                }
            }
        }
    }
}

// ---------------------------------------------------------------------------
// RoPE, in-place on Q16/K16 (head-major [h][t][d]).
// ---------------------------------------------------------------------------
__global__ void __launch_bounds__(256)
rope_kernel(_Float16* __restrict__ Q16, _Float16* __restrict__ K16,
            const int* __restrict__ positions)
{
    int idx = blockIdx.x * blockDim.x + threadIdx.x;   // TT * NH * 64
    int dd = idx & 63;
    int h  = (idx >> 6) & 15;
    int t  = idx >> 10;
    float invf = __expf(-0.14391156f * (float)dd);     // 10000^(-dd/64)
    float fr = (float)positions[t] * invf;
    float s, c;
    __sincosf(fr, &s, &c);
    size_t base = (size_t)(h * TT + t) * HD;

    float q1 = (float)Q16[base + dd], q2 = (float)Q16[base + 64 + dd];
    Q16[base + dd]      = (_Float16)(q1 * c - q2 * s);
    Q16[base + 64 + dd] = (_Float16)(q2 * c + q1 * s);

    float k1 = (float)K16[base + dd], k2 = (float)K16[base + 64 + dd];
    K16[base + dd]      = (_Float16)(k1 * c - k2 * s);
    K16[base + 64 + dd] = (_Float16)(k2 * c + k1 * s);
}

// ---------------------------------------------------------------------------
// Causal flash attention: 1 wave per 16-row q tile, 4 waves/block.
// Output f16 head-interleaved [t][2048] for the following f16 GEMM.
// ---------------------------------------------------------------------------
__global__ void __launch_bounds__(128)
attn_kernel(const _Float16* __restrict__ Q16, const _Float16* __restrict__ K16,
            const _Float16* __restrict__ V16, _Float16* __restrict__ O16)
{
    __shared__ _Float16 lds[4][HD * 32 + 16 * 32];   // per wave: Vt[128][32] + P[16][32]

    const int tid  = threadIdx.x;
    const int w    = tid >> 5;
    const int lane = tid & 31;
    const int l16  = lane & 15;
    const int hi   = lane >> 4;
    const int head = blockIdx.x;
    const int qb   = (blockIdx.y * 4 + w) * 16;

    const _Float16* Qh = Q16 + (size_t)head * TT * HD;
    const _Float16* Kh = K16 + (size_t)head * TT * HD;
    const _Float16* Vh = V16 + (size_t)head * TT * HD;
    _Float16* Vt = &lds[w][0];
    _Float16* Pl = &lds[w][HD * 32];

    v16h qf[4];
#pragma unroll
    for (int kc = 0; kc < 4; kc++) {
        const _Float16* p = &Qh[(size_t)(qb + l16) * HD + kc * 32];
        qf[kc] = pk16(*reinterpret_cast<const v8h*>(p + 16 * hi),
                      *reinterpret_cast<const v8h*>(p + 16 * hi + 8));
    }

    v8f o[8];
    float mx[8], sm[8];
#pragma unroll
    for (int dc = 0; dc < 8; dc++)
#pragma unroll
        for (int r = 0; r < 8; r++) o[dc][r] = 0.0f;
#pragma unroll
    for (int r = 0; r < 8; r++) { mx[r] = -1e30f; sm[r] = 0.0f; }

    const float scale = 0.0883883476f;   // 1/sqrt(128)

    for (int kb = 0; kb < qb + 16; kb += 32) {
        if (kb + 32 < TT) {
            __builtin_prefetch(&Kh[(size_t)(kb + 32 + lane) * HD], 0, 0);
            __builtin_prefetch(&Vh[(size_t)(kb + 32 + lane) * HD], 0, 0);
        }
        // stage V tile transposed: Vt[d][key], key = lane
#pragma unroll
        for (int j = 0; j < 16; j++) {
            v8h t8 = *reinterpret_cast<const v8h*>(&Vh[(size_t)(kb + lane) * HD + j * 8]);
#pragma unroll
            for (int e = 0; e < 8; e++) Vt[(j * 8 + e) * 32 + lane] = t8[e];
        }

        v8f s0, s1;
#pragma unroll
        for (int r = 0; r < 8; r++) { s0[r] = 0.0f; s1[r] = 0.0f; }
#pragma unroll
        for (int kc = 0; kc < 4; kc++) {
            const _Float16* kp0 = &Kh[(size_t)(kb + l16) * HD + kc * 32 + 16 * hi];
            s0 = wmma_f16(qf[kc],
                          pk16(*reinterpret_cast<const v8h*>(kp0),
                               *reinterpret_cast<const v8h*>(kp0 + 8)), s0);
            const _Float16* kp1 = &Kh[(size_t)(kb + 16 + l16) * HD + kc * 32 + 16 * hi];
            s1 = wmma_f16(qf[kc],
                          pk16(*reinterpret_cast<const v8h*>(kp1),
                               *reinterpret_cast<const v8h*>(kp1 + 8)), s1);
        }

        const int key0 = kb + l16, key1 = kb + 16 + l16;
#pragma unroll
        for (int r = 0; r < 8; r++) {
            int q = qb + r + 8 * hi;
            float a = s0[r] * scale; if (key0 > q) a = -1e30f;
            float b = s1[r] * scale; if (key1 > q) b = -1e30f;
            float tm = fmaxf(a, b);
#pragma unroll
            for (int off = 1; off < 16; off <<= 1)
                tm = fmaxf(tm, __shfl_xor(tm, off, 32));
            float nm = fmaxf(mx[r], tm);
            float cf = __expf(mx[r] - nm);
            mx[r] = nm;
            float p0 = __expf(a - nm);
            float p1 = __expf(b - nm);
            float ps = p0 + p1;
#pragma unroll
            for (int off = 1; off < 16; off <<= 1)
                ps += __shfl_xor(ps, off, 32);
            sm[r] = sm[r] * cf + ps;
#pragma unroll
            for (int dc = 0; dc < 8; dc++) o[dc][r] *= cf;
            Pl[(r + 8 * hi) * 32 + l16]      = (_Float16)p0;
            Pl[(r + 8 * hi) * 32 + 16 + l16] = (_Float16)p1;
        }

        v16h af = pk16(*reinterpret_cast<const v8h*>(&Pl[l16 * 32 + 16 * hi]),
                       *reinterpret_cast<const v8h*>(&Pl[l16 * 32 + 16 * hi + 8]));
#pragma unroll
        for (int dc = 0; dc < 8; dc++) {
            const _Float16* vp = &Vt[(dc * 16 + l16) * 32 + 16 * hi];
            o[dc] = wmma_f16(af,
                             pk16(*reinterpret_cast<const v8h*>(vp),
                                  *reinterpret_cast<const v8h*>(vp + 8)), o[dc]);
        }
    }

#pragma unroll
    for (int dc = 0; dc < 8; dc++)
#pragma unroll
        for (int r = 0; r < 8; r++) {
            int t = qb + r + 8 * hi;
            O16[(size_t)t * HID + head * HD + dc * 16 + l16] =
                (_Float16)(o[dc][r] / sm[r]);
        }
}

// ---------------------------------------------------------------------------
extern "C" void kernel_launch(void* const* d_in, const int* in_sizes, int n_in,
                              void* d_out, int out_size, void* d_ws, size_t ws_size,
                              hipStream_t stream)
{
    const float* hidden    = (const float*)d_in[0];   // [T][2048]
    const int*   positions = (const int*)d_in[1];     // [T]
    const float* w_qkv     = (const float*)d_in[2];   // [2048][6144]
    const float* w_o       = (const float*)d_in[3];   // [2048][2048]
    float* out = (float*)d_out;                        // [T][2048] f32

    // workspace: A16 | W1t | W2t | Q16 | K16 | V16 | O16  (all f16)
    _Float16* A16 = (_Float16*)d_ws;                        // T*H
    _Float16* W1t = A16 + (size_t)TT * HID;                 // 3H*H  ([n][k])
    _Float16* W2t = W1t + (size_t)3 * HID * HID;            // H*H   ([n][k])
    _Float16* Q16 = W2t + (size_t)HID * HID;                // NH*T*HD
    _Float16* K16 = Q16 + (size_t)TT * HID;
    _Float16* V16 = K16 + (size_t)TT * HID;
    _Float16* O16 = V16 + (size_t)TT * HID;                 // T*H

    // 0) one-shot f16 conversion (+ weight transpose to [n][k])
    cvt_f16_kernel<<<(TT * HID / 4 + 255) / 256, 256, 0, stream>>>(hidden, A16, TT * HID / 4);
    tcvt_kernel<<<dim3(3 * HID / 32, HID / 32), 256, 0, stream>>>(w_qkv, W1t, HID, 3 * HID);
    tcvt_kernel<<<dim3(HID / 32, HID / 32), 256, 0, stream>>>(w_o, W2t, HID, HID);

    // 1) QKV projection -> f16 head-major Q/K/V
    gemm_wmma_kernel<<<dim3(3 * HID / TILE_N, TT / TILE_M), 256, 0, stream>>>(
        A16, W1t, nullptr, Q16, K16, V16, TT, 3 * HID, HID, 1);

    // 2) RoPE in place on Q,K
    rope_kernel<<<(TT * NH * 64) / 256, 256, 0, stream>>>(Q16, K16, positions);

    // 3) causal flash attention -> f16
    attn_kernel<<<dim3(NH, TT / 64), 128, 0, stream>>>(Q16, K16, V16, O16);

    // 4) output projection -> f32 out
    gemm_wmma_kernel<<<dim3(HID / TILE_N, TT / TILE_M), 256, 0, stream>>>(
        O16, W2t, out, nullptr, nullptr, nullptr, TT, HID, HID, 0);
}
